// NET_51908974739547
// MI455X (gfx1250) — compile-verified, matrix operand order checked by reading, not run on hardware
//
#include <hip/hip_runtime.h>
#include <math.h>

// ---------------- types ----------------
typedef __bf16 bf16_t;
typedef bf16_t v16bf __attribute__((ext_vector_type(16)));
typedef bf16_t v8bf  __attribute__((ext_vector_type(8)));
typedef float  v8f   __attribute__((ext_vector_type(8)));

// ---------------- small utility kernels ----------------
__global__ void k_zero_f32(float* __restrict__ p, int n) {
    int i = blockIdx.x * blockDim.x + threadIdx.x;
    if (i < n) p[i] = 0.0f;
}

__global__ void k_deg_count(const int* __restrict__ col, float* __restrict__ deg, int E) {
    int e = blockIdx.x * blockDim.x + threadIdx.x;
    if (e < E) atomicAdd(&deg[col[e]], 1.0f);
}

__global__ void k_dinv(float* __restrict__ deg, int n) {
    int i = blockIdx.x * blockDim.x + threadIdx.x;
    if (i < n) deg[i] = rsqrtf(deg[i] + 1.0f);  // +1 self loop
}

// fp32 -> bf16 convert, optional fused ReLU
__global__ void k_cvt_bf16(const float* __restrict__ src, unsigned short* __restrict__ dst,
                           long n, int relu) {
    long i = (long)blockIdx.x * blockDim.x + threadIdx.x;
    if (i >= n) return;
    float v = src[i];
    if (relu) v = fmaxf(v, 0.0f);
    ((bf16_t*)dst)[i] = (bf16_t)v;
}

// out[i,:] = dinv[i]^2 * h[i,:]   (self-loop term; initializes the accumulator)
__global__ void k_self_init(const float* __restrict__ h, const float* __restrict__ dinv,
                            float* __restrict__ out, long total4, int shift /*log2(d/4)*/) {
    long idx = (long)blockIdx.x * blockDim.x + threadIdx.x;
    if (idx >= total4) return;
    long i = idx >> shift;
    float s = dinv[i]; s *= s;
    float4 v = ((const float4*)h)[idx];
    float4 o = { s * v.x, s * v.y, s * v.z, s * v.w };
    ((float4*)out)[idx] = o;
}

// per (edge, float4-chunk): gather h[row], scale by dinv[row]*dinv[col], atomic-add to out[col]
__global__ void k_edge_scatter(const int* __restrict__ row, const int* __restrict__ col,
                               const float* __restrict__ dinv, const float* __restrict__ h,
                               float* __restrict__ out, long total4, int shift /*log2(d/4)*/) {
    long idx = (long)blockIdx.x * blockDim.x + threadIdx.x;
    if (idx >= total4) return;
    long e  = idx >> shift;
    int  c4 = (int)(idx & ((1 << shift) - 1));
    int  r  = row[e];
    int  c  = col[e];
    float nrm = dinv[r] * dinv[c];
    float4 v = ((const float4*)h)[((long)r << shift) + c4];
    float* o = out + ((((long)c << shift) + c4) << 2);
    atomicAdd(o + 0, nrm * v.x);   // non-returning GLOBAL_ATOMIC_ADD_F32
    atomicAdd(o + 1, nrm * v.y);
    atomicAdd(o + 2, nrm * v.z);
    atomicAdd(o + 3, nrm * v.w);
}

// ---------------- WMMA GEMM: C[n,NO] = A[n,K](bf16) @ B[NO,K](bf16)^T + bias ----------------
// One wave computes one 16x16 tile; block = (NO/16) waves spanning all output columns of a
// 16-row node tile. N is an exact multiple of 16 (100000 = 6250*16) so EXEC is all ones.
__device__ __forceinline__ v16bf load_frag16x32(const bf16_t* __restrict__ rowptr,
                                                int kk, int lane) {
    // ISA 7.12.2: 16-bit A/B fragment: lanes {0-15,16-31} hold rows M(N)=lane%16;
    // elems 0..7 = K kbase..kbase+7, elems 8..15 = K kbase+16..kbase+23,
    // kbase = 32*kk + (lane>=16 ? 8 : 0).
    int kbase = (kk << 5) + ((lane & 16) ? 8 : 0);
    v8bf lo = *(const v8bf*)(rowptr + kbase);        // 16B aligned (K mult of 8 elems)
    v8bf hi = *(const v8bf*)(rowptr + kbase + 16);
    v16bf f;
#pragma unroll
    for (int i = 0; i < 8; ++i) { f[i] = lo[i]; f[8 + i] = hi[i]; }
    return f;
}

__global__ void __launch_bounds__(256)
k_gemm_wmma_bf16(const unsigned short* __restrict__ Au, const unsigned short* __restrict__ Bu,
                 const float* __restrict__ bias, float* __restrict__ C, int K, int NO) {
    const bf16_t* A = (const bf16_t*)Au;
    const bf16_t* B = (const bf16_t*)Bu;
    int lane = threadIdx.x & 31;
    int wave = threadIdx.x >> 5;
    int rowBase = blockIdx.x << 4;             // 16 nodes per block
    int colBase = wave << 4;                   // 16 output features per wave
    int rIdx = rowBase + (lane & 15);
    int cIdx = colBase + (lane & 15);

    const bf16_t* arow = A + (long)rIdx * K;
    const bf16_t* brow = B + (long)cIdx * K;   // B stored [NO,K] row-major == Bᵀ fragment

    v8f acc = {};
    int ksteps = K >> 5;
#pragma unroll
    for (int kk = 0; kk < 4; ++kk) {
        if (kk >= ksteps) break;
        v16bf a = load_frag16x32(arow, kk, lane);
        v16bf b = load_frag16x32(brow, kk, lane);
        acc = __builtin_amdgcn_wmma_f32_16x16x32_bf16(
            /*neg_a=*/false, a, /*neg_b=*/false, b,
            /*c_mod=*/(short)0, acc, /*reuse_a=*/false, /*reuse_b=*/false);
    }

    float bv = bias[cIdx];
    int mBase = rowBase + ((lane & 16) ? 8 : 0);   // C/D layout: VGPR v -> M = v (+8 hi lanes)
#pragma unroll
    for (int v = 0; v < 8; ++v) {
        C[(long)(mBase + v) * NO + cIdx] = acc[v] + bv;
    }
}

// ---------------- orchestration ----------------
static inline unsigned nblk(long total, int b) { return (unsigned)((total + b - 1) / b); }

extern "C" void kernel_launch(void* const* d_in, const int* in_sizes, int n_in,
                              void* d_out, int out_size, void* d_ws, size_t ws_size,
                              hipStream_t stream) {
    const float* x    = (const float*)d_in[0];
    const int*   ei   = (const int*)  d_in[1];
    const float* W1   = (const float*)d_in[2];
    const float* b1   = (const float*)d_in[3];
    const float* W2   = (const float*)d_in[4];
    const float* b2   = (const float*)d_in[5];
    const float* Wout = (const float*)d_in[6];
    const float* bout = (const float*)d_in[7];
    const float* Wlin = (const float*)d_in[8];
    const float* blin = (const float*)d_in[9];

    const int DIN = 128, DH = 128, DO = 64, DL = 64;
    int n = in_sizes[0] / DIN;           // 100000 (multiple of 16)
    int E = in_sizes[1] / 2;             // 1600000
    const int* row = ei;
    const int* col = ei + E;

    // workspace carve-up
    char* ws = (char*)d_ws;
    float*          dinv = (float*)ws;                               // n f32 (deg -> dinv in place)
    unsigned short* w1b  = (unsigned short*)(ws + (512u << 10));     // 128*128 bf16
    unsigned short* w2b  = w1b + DH * DIN;
    unsigned short* wob  = w2b + DH * DH;                            // 64*128
    unsigned short* wlb  = wob + DO * DH;                            // 64*64
    unsigned short* xbf  = (unsigned short*)(ws + (1u  << 20));      // n*128 bf16 (25.6 MB)
    float*          hbuf = (float*)(ws + (28u << 20));               // n*128 f32  (51.2 MB)
    float*          abuf = (float*)(ws + (80u << 20));               // n*128 f32  (51.2 MB)

    const int B = 256;

    // --- degrees -> dinv ---
    k_zero_f32 <<<nblk(n, B), B, 0, stream>>>(dinv, n);
    k_deg_count<<<nblk(E, B), B, 0, stream>>>(col, dinv, E);
    k_dinv     <<<nblk(n, B), B, 0, stream>>>(dinv, n);

    // --- weights -> bf16 ---
    k_cvt_bf16<<<nblk(DH * DIN, B), B, 0, stream>>>(W1,   w1b, DH * DIN, 0);
    k_cvt_bf16<<<nblk(DH * DH,  B), B, 0, stream>>>(W2,   w2b, DH * DH,  0);
    k_cvt_bf16<<<nblk(DO * DH,  B), B, 0, stream>>>(Wout, wob, DO * DH,  0);
    k_cvt_bf16<<<nblk(DL * DO,  B), B, 0, stream>>>(Wlin, wlb, DL * DO,  0);

    long nd128 = (long)n * 128, nd64 = (long)n * 64;
    long n4_128 = (long)n * 32,  n4_64 = (long)n * 16;     // float4 counts
    long e4_128 = (long)E * 32,  e4_64 = (long)E * 16;

    // --- layer 1: h = x@W1ᵀ+b1 ; agg ; relu fused into next cvt ---
    k_cvt_bf16     <<<nblk(nd128, B), B, 0, stream>>>(x, xbf, nd128, 0);
    k_gemm_wmma_bf16<<<n / 16, 256, 0, stream>>>(xbf, w1b, b1, hbuf, DIN, DH);
    k_self_init    <<<nblk(n4_128, B), B, 0, stream>>>(hbuf, dinv, abuf, n4_128, 5);
    k_edge_scatter <<<nblk(e4_128, B), B, 0, stream>>>(row, col, dinv, hbuf, abuf, e4_128, 5);

    // --- layer 2 ---
    k_cvt_bf16     <<<nblk(nd128, B), B, 0, stream>>>(abuf, xbf, nd128, 1 /*relu*/);
    k_gemm_wmma_bf16<<<n / 16, 256, 0, stream>>>(xbf, w2b, b2, hbuf, DH, DH);
    k_self_init    <<<nblk(n4_128, B), B, 0, stream>>>(hbuf, dinv, abuf, n4_128, 5);
    k_edge_scatter <<<nblk(e4_128, B), B, 0, stream>>>(row, col, dinv, hbuf, abuf, e4_128, 5);

    // --- layer 3 (out dim 64, no relu after agg) ---
    k_cvt_bf16     <<<nblk(nd128, B), B, 0, stream>>>(abuf, xbf, nd128, 1 /*relu*/);
    k_gemm_wmma_bf16<<<n / 16, 128, 0, stream>>>(xbf, wob, bout, hbuf, DH, DO);
    k_self_init    <<<nblk(n4_64, B), B, 0, stream>>>(hbuf, dinv, abuf, n4_64, 4);
    k_edge_scatter <<<nblk(e4_64, B), B, 0, stream>>>(row, col, dinv, hbuf, abuf, e4_64, 4);

    // --- final linear: d_out = agg3@Wlinᵀ + blin ---
    k_cvt_bf16     <<<nblk(nd64, B), B, 0, stream>>>(abuf, xbf, nd64, 0);
    k_gemm_wmma_bf16<<<n / 16, 128, 0, stream>>>(xbf, wlb, blin, (float*)d_out, DO, DL);
}
